// Probabilistic_NetworkDiffusion_88622355186389
// MI455X (gfx1250) — matrix operand order repaired and checked by reading, not compiled, not run on hardware
//
#include <hip/hip_runtime.h>
#include <hip/hip_bf16.h>

#define T_DIM 64
#define S_DIM 1024
#define D_DIM 2048
// 1/sqrt(2048)
#define NORM_INV 0.022097086912079608f

typedef __attribute__((ext_vector_type(16))) __bf16         v16bf;
typedef __attribute__((ext_vector_type(8)))  float          v8f;
typedef __attribute__((ext_vector_type(4)))  float          fvec4;
typedef __attribute__((ext_vector_type(4)))  unsigned int   v4u;
typedef __attribute__((ext_vector_type(4)))  unsigned short v4us;

// Exact pointee type the async builtin expects (per hipcc diagnostic):
// gcc-style vector_size(16) int, addrspace(1) source / addrspace(3) dest.
typedef int v4i_vs __attribute__((vector_size(16)));
typedef __attribute__((address_space(1))) v4i_vs* gptr_b128;
typedef __attribute__((address_space(3))) v4i_vs* lptr_b128;

// Fragment register image: q[0] -> VGPR0..3, q[1] -> VGPR4..7 of the bf16 operand
union Frag16 { v16bf v; v4u q[2]; };

__device__ __forceinline__ unsigned short f32_to_bf16_rn(float f) {
    unsigned int u = __float_as_uint(f);
    u += 0x7fffu + ((u >> 16) & 1u);          // round-to-nearest-even
    return (unsigned short)(u >> 16);
}
__device__ __forceinline__ float bf16_to_f32(unsigned short h) {
    return __uint_as_float(((unsigned int)h) << 16);
}

// CDNA5 async global->LDS copy (ASYNCcnt path), with synchronous b128 fallback.
#if __has_builtin(__builtin_amdgcn_global_load_async_to_lds_b128)
#define HAVE_ASYNC_COPY 1
#else
#define HAVE_ASYNC_COPY 0
#endif

__device__ __forceinline__ void copy16_g2l(const unsigned short* __restrict__ g,
                                           unsigned short* __restrict__ l) {
#if HAVE_ASYNC_COPY
    __builtin_amdgcn_global_load_async_to_lds_b128((gptr_b128)g, (lptr_b128)l, 0, 0);
#else
    *(v4u*)l = *(const v4u*)g;
#endif
}
__device__ __forceinline__ void wait_async_copies() {
#if HAVE_ASYNC_COPY
#if __has_builtin(__builtin_amdgcn_s_wait_asynccnt)
    __builtin_amdgcn_s_wait_asynccnt(0);
#else
    asm volatile("s_wait_asynccnt 0" ::: "memory");
#endif
#endif
}

// LDS row stride: 40 shorts = 80 B. 16B-aligned rows (ds_load_b128-able) and
// 20-dword pitch => 16 consecutive rows hit 16 disjoint 4-bank windows.
#define LDSP 40

// ---------------------------------------------------------------------------
// Prep A: split eps into bf16 hi/lo planes (done ONCE instead of per n-block).
// ---------------------------------------------------------------------------
__global__ void __launch_bounds__(256)
prep_eps_kernel(const float* __restrict__ eps, unsigned short* __restrict__ ehi,
                unsigned short* __restrict__ elo)
{
    const size_t base = ((size_t)blockIdx.x * 256 + threadIdx.x) * 4;
    const fvec4 f = *(const fvec4*)&eps[base];
    v4us hi, lo;
    #pragma unroll
    for (int j = 0; j < 4; ++j) {
        const unsigned short h = f32_to_bf16_rn(f[j]);
        hi[j] = h;
        lo[j] = f32_to_bf16_rn(f[j] - bf16_to_f32(h));
    }
    *(v4us*)&ehi[base] = hi;
    *(v4us*)&elo[base] = lo;
}

// ---------------------------------------------------------------------------
// Prep B: cholT[n][k] = (k > n) ? Psi[k][n] : 0, as bf16 hi/lo planes,
// TRANSPOSED so the GEMM B-staging is a contiguous copy. Mask + transpose done
// once instead of per m-block. Tiles never touched by the triangular GEMM
// (k0+32 <= floor(n0/64)*64) are skipped entirely.
// ---------------------------------------------------------------------------
__global__ void __launch_bounds__(256)
prep_cholT_kernel(const float* __restrict__ Psi, unsigned short* __restrict__ chT_hi,
                  unsigned short* __restrict__ chT_lo)
{
    const int k0 = blockIdx.x * 32;
    const int n0 = blockIdx.y * 32;
    if (k0 + 32 <= (n0 & ~63)) return;      // uniform early-out: GEMM never reads it

    __shared__ unsigned short th[32 * 33];
    __shared__ unsigned short tl[32 * 33];
    const int tid = threadIdx.x;
    {
        const int kl = tid >> 3, nl4 = (tid & 7) * 4;
        const int kg = k0 + kl;
        const fvec4 f = *(const fvec4*)&Psi[(size_t)kg * D_DIM + n0 + nl4];
        #pragma unroll
        for (int j = 0; j < 4; ++j) {
            const float v = (kg > n0 + nl4 + j) ? f[j] : 0.0f;
            const unsigned short h = f32_to_bf16_rn(v);
            th[kl * 33 + nl4 + j] = h;
            tl[kl * 33 + nl4 + j] = f32_to_bf16_rn(v - bf16_to_f32(h));
        }
    }
    __syncthreads();
    {
        const int nl = tid >> 3, kl4 = (tid & 7) * 4;
        v4us hi, lo;
        #pragma unroll
        for (int j = 0; j < 4; ++j) {
            hi[j] = th[(kl4 + j) * 33 + nl];
            lo[j] = tl[(kl4 + j) * 33 + nl];
        }
        *(v4us*)&chT_hi[(size_t)(n0 + nl) * D_DIM + k0 + kl4] = hi;
        *(v4us*)&chT_lo[(size_t)(n0 + nl) * D_DIM + k0 + kl4] = lo;
    }
}

// ---------------------------------------------------------------------------
// Kernel 1: phi[s,n] = exp((eps @ (tril(Psi,-1)+I) + nu)/sqrt(D))
// Triangular GEMM (K-blocks below n0 skipped); unit diagonal folded into the
// epilogue as +eps[s,n]. bf16x3 split on v_wmma_f32_16x16x32_bf16 gives
// ~fp32-class accuracy at 3 WMMAs per tile pair.
// Block tile 64x64, BK=32, 256 threads = 8 waves (4x2). Staging is four pure
// 16B global->LDS copies per thread (async if available).
// ---------------------------------------------------------------------------
__global__ void __launch_bounds__(256)
gemm_phi_kernel(const unsigned short* __restrict__ ehi, const unsigned short* __restrict__ elo,
                const unsigned short* __restrict__ chT_hi, const unsigned short* __restrict__ chT_lo,
                const float* __restrict__ eps, const float* __restrict__ nu,
                float* __restrict__ out_phi)
{
    __shared__ unsigned short sAhi[64 * LDSP];
    __shared__ unsigned short sAlo[64 * LDSP];
    __shared__ unsigned short sBhi[64 * LDSP];
    __shared__ unsigned short sBlo[64 * LDSP];

    const int tid  = threadIdx.x;
    const int lane = tid & 31;
    const int wave = tid >> 5;
    const int wm   = wave >> 1;      // 0..3 -> rows wm*16
    const int wn   = wave & 1;       // 0..1 -> cols wn*32
    const int n0   = blockIdx.x * 64;
    const int m0   = blockIdx.y * 64;

    // staging map: one 16B chunk per thread per plane
    const int r  = tid >> 2;         // 0..63
    const int c8 = (tid & 3) * 8;    // 0,8,16,24

    v8f acc[2] = {};

    for (int kb = n0; kb < D_DIM; kb += 32) {
        // branchless prefetch of next K-block (global_prefetch_b8)
        {
            const int kpf = (kb + 32 < D_DIM) ? (kb + 32) : kb;
            __builtin_prefetch(&ehi[(size_t)(m0 + r) * D_DIM + kpf + c8], 0, 3);
            __builtin_prefetch(&elo[(size_t)(m0 + r) * D_DIM + kpf + c8], 0, 3);
            __builtin_prefetch(&chT_hi[(size_t)(n0 + r) * D_DIM + kpf + c8], 0, 3);
            __builtin_prefetch(&chT_lo[(size_t)(n0 + r) * D_DIM + kpf + c8], 0, 3);
        }

        // stage tiles: pure b128 copies, layouts already fragment-ready
        copy16_g2l(&ehi[(size_t)(m0 + r) * D_DIM + kb + c8], &sAhi[r * LDSP + c8]);
        copy16_g2l(&elo[(size_t)(m0 + r) * D_DIM + kb + c8], &sAlo[r * LDSP + c8]);
        copy16_g2l(&chT_hi[(size_t)(n0 + r) * D_DIM + kb + c8], &sBhi[r * LDSP + c8]);
        copy16_g2l(&chT_lo[(size_t)(n0 + r) * D_DIM + kb + c8], &sBlo[r * LDSP + c8]);
        wait_async_copies();
        __syncthreads();

        // A fragments: 16-bit A-matrix 16x32 wave32 layout; two 16B runs.
        Frag16 ahi, alo;
        {
            const int base = (wm * 16 + (lane & 15)) * LDSP + ((lane >> 4) << 3);
            ahi.q[0] = *(const v4u*)&sAhi[base];
            ahi.q[1] = *(const v4u*)&sAhi[base + 16];
            alo.q[0] = *(const v4u*)&sAlo[base];
            alo.q[1] = *(const v4u*)&sAlo[base + 16];
        }
        // B fragments + 3 WMMAs per 16x16 N-tile (bf16x3 compensation).
        #pragma unroll
        for (int nt = 0; nt < 2; ++nt) {
            Frag16 bhi, blo;
            const int base = (wn * 32 + nt * 16 + (lane & 15)) * LDSP + ((lane >> 4) << 3);
            bhi.q[0] = *(const v4u*)&sBhi[base];
            bhi.q[1] = *(const v4u*)&sBhi[base + 16];
            blo.q[0] = *(const v4u*)&sBlo[base];
            blo.q[1] = *(const v4u*)&sBlo[base + 16];
            acc[nt] = __builtin_amdgcn_wmma_f32_16x16x32_bf16(
                false, ahi.v, false, bhi.v, (short)0, acc[nt], false, false);
            acc[nt] = __builtin_amdgcn_wmma_f32_16x16x32_bf16(
                false, ahi.v, false, blo.v, (short)0, acc[nt], false, false);
            acc[nt] = __builtin_amdgcn_wmma_f32_16x16x32_bf16(
                false, alo.v, false, bhi.v, (short)0, acc[nt], false, false);
        }
        __syncthreads();
    }

    // epilogue: + eps (unit diagonal), + nu, scale, exp, store phi
    #pragma unroll
    for (int nt = 0; nt < 2; ++nt) {
        const int n = n0 + wn * 32 + nt * 16 + (lane & 15);
        const float nun = nu[n];
        #pragma unroll
        for (int rr = 0; rr < 8; ++rr) {
            const int ml = wm * 16 + ((lane < 16) ? rr : (8 + rr));
            const int s  = m0 + ml;
            const float pre = acc[nt][rr] + eps[(size_t)s * D_DIM + n] + nun;
            out_phi[(size_t)s * D_DIM + n] = __expf(pre * NORM_INV);
        }
    }
}

// ---------------------------------------------------------------------------
// Kernel 2: per-row reductions  ws[s] = ||phi_s||^2 , ws[S+s] = phi_s . x0
// ---------------------------------------------------------------------------
__global__ void __launch_bounds__(256)
row_reduce_kernel(const float* __restrict__ phi, const float* __restrict__ x0,
                  float* __restrict__ ws)
{
    const int s   = blockIdx.x;
    const int tid = threadIdx.x;
    const fvec4* row = (const fvec4*)(phi + (size_t)s * D_DIM);
    const fvec4* xv  = (const fvec4*)x0;
    float sq = 0.0f, xp = 0.0f;
    #pragma unroll
    for (int i = 0; i < 2; ++i) {
        const int idx = tid + i * 256;
        const fvec4 p = row[idx];
        const fvec4 x = xv[idx];
        sq += p.x * p.x + p.y * p.y + p.z * p.z + p.w * p.w;
        xp += p.x * x.x + p.y * x.y + p.z * x.z + p.w * x.w;
    }
    __shared__ float s_sq[256];
    __shared__ float s_xp[256];
    s_sq[tid] = sq; s_xp[tid] = xp;
    __syncthreads();
    for (int off = 128; off > 0; off >>= 1) {
        if (tid < off) { s_sq[tid] += s_sq[tid + off]; s_xp[tid] += s_xp[tid + off]; }
        __syncthreads();
    }
    if (tid == 0) { ws[s] = s_sq[0]; ws[S_DIM + s] = s_xp[0]; }
}

// ---------------------------------------------------------------------------
// Kernel 3: x_t_pred[t,s,:] = decay*x0phi/row_sq * phi[s,:]  (512 MB NT stream)
//           x0xt_pred[t,s]  = (decay*x0phi)^2 / row_sq  (einsum collapses:
//           dot(fea_vec[s], phi[s]) == x0phi[s] => diag = decay*x0phi)
// phi (8 MB) stays L2-resident across all 64 t-slices; nontemporal stores keep
// the 512 MB stream from evicting it.
// ---------------------------------------------------------------------------
__global__ void __launch_bounds__(256)
expand_kernel(const float* __restrict__ phi, const float* __restrict__ time,
              const float* __restrict__ ws, float* __restrict__ out_xt,
              float* __restrict__ out_x0xt)
{
    const int s   = blockIdx.x;
    const int t   = blockIdx.y;
    const int tid = threadIdx.x;
    const float rs    = ws[s];
    const float xp    = ws[S_DIM + s];
    const float decay = __expf(-time[t] * rs);       // BETA = 1
    const float c1    = decay * xp / rs;

    const fvec4* prow = (const fvec4*)(phi + (size_t)s * D_DIM);
    fvec4* orow = (fvec4*)(out_xt + ((size_t)t * S_DIM + s) * D_DIM);
    #pragma unroll
    for (int i = 0; i < 2; ++i) {
        const int idx = tid + i * 256;
        fvec4 p = prow[idx];
        fvec4 o = p * c1;
        __builtin_nontemporal_store(o, &orow[idx]);
    }
    if (tid == 0) {
        const float diag = decay * xp;
        out_x0xt[(size_t)t * S_DIM + s] = diag * diag / rs;
    }
}

// ---------------------------------------------------------------------------
extern "C" void kernel_launch(void* const* d_in, const int* in_sizes, int n_in,
                              void* d_out, int out_size, void* d_ws, size_t ws_size,
                              hipStream_t stream)
{
    const float* time = (const float*)d_in[0];
    const float* x0   = (const float*)d_in[1];
    const float* nu   = (const float*)d_in[2];
    const float* Psi  = (const float*)d_in[3];
    const float* eps  = (const float*)d_in[4];

    float* out      = (float*)d_out;
    float* out_xt   = out;                                              // [T,S,D]
    float* out_x0xt = out + (size_t)T_DIM * S_DIM * D_DIM;              // [T,S]
    float* out_phi  = out_x0xt + (size_t)T_DIM * S_DIM;                 // [S,D]
    float* ws       = (float*)d_ws;                                     // 2*S floats

    // bf16 hi/lo planes live in the head of the x_t_pred region (written only
    // by expand_kernel at the very end) -> 25 MB of free, deterministic scratch.
    unsigned short* ehi    = (unsigned short*)out_xt;
    unsigned short* elo    = ehi    + (size_t)S_DIM * D_DIM;
    unsigned short* chT_hi = elo    + (size_t)S_DIM * D_DIM;
    unsigned short* chT_lo = chT_hi + (size_t)D_DIM * D_DIM;

    prep_eps_kernel<<<(S_DIM * D_DIM) / (4 * 256), 256, 0, stream>>>(eps, ehi, elo);
    prep_cholT_kernel<<<dim3(D_DIM / 32, D_DIM / 32), 256, 0, stream>>>(Psi, chT_hi, chT_lo);
    gemm_phi_kernel<<<dim3(D_DIM / 64, S_DIM / 64), 256, 0, stream>>>(
        ehi, elo, chT_hi, chT_lo, eps, nu, out_phi);
    row_reduce_kernel<<<S_DIM, 256, 0, stream>>>(out_phi, x0, ws);
    expand_kernel<<<dim3(S_DIM, T_DIM), 256, 0, stream>>>(out_phi, time, ws, out_xt, out_x0xt);
}
